// ShortGINE_88630945120295
// MI455X (gfx1250) — matrix operand order repaired and checked by reading, not compiled, not run on hardware
//
#include <hip/hip_runtime.h>

// ---------------------------------------------------------------------------
// GINE layer for MI455X (gfx1250, wave32).
// Memory-bound problem (~1.1GB traffic vs ~36 GFLOP) -> f16 WMMA GEMMs with
// f32 accumulation, weights staged in LDS in B-layout ONCE per block
// (persistent wave-level grid-stride loop over tiles), gather/scatter fused
// into the edge GEMM epilogue, RMSNorm fused via intra-half shuffles.
// ---------------------------------------------------------------------------

#define N_NODES 100000
#define N_EDGES 600000
#define DIM     128
#define EDIM    64
#define RMS_EPS 1.1920929e-07f

#define EDGE_GRID 1250   // persistent blocks for edge kernel (4 waves each)
#define NODE_GRID 625    // persistent blocks for node kernels

typedef __attribute__((ext_vector_type(16))) _Float16 v16h;
typedef __attribute__((ext_vector_type(8)))  _Float16 v8h;
typedef __attribute__((ext_vector_type(8)))  float    v8f;
typedef __attribute__((ext_vector_type(4)))  float    v4f;

static __device__ __forceinline__ v8f wmma16(v16h a, v16h b, v8f c) {
  // v_wmma_f32_16x16x32_f16: D = A(16x32 f16) * B(32x16 f16) + C(16x16 f32)
  return __builtin_amdgcn_wmma_f32_16x16x32_f16(false, a, false, b, (short)0, c,
                                                false, false);
}

// Build a v16h from two contiguous groups of 8 halves (LDS or global).
static __device__ __forceinline__ v16h pack16(const _Float16* p0, const _Float16* p1) {
  v8h lo = *(const v8h*)p0;
  v8h hi = *(const v8h*)p1;
  v16h r;
#pragma unroll
  for (int i = 0; i < 8; ++i) { r[i] = lo[i]; r[i + 8] = hi[i]; }
  return r;
}

// A-layout pack from a row of f32: halves = p[kb+0..7], p[kb+16..23]
static __device__ __forceinline__ v16h packA_f32(const float* p, int kb) {
  v4f a0 = *(const v4f*)(p + kb);
  v4f a1 = *(const v4f*)(p + kb + 4);
  v4f b0 = *(const v4f*)(p + kb + 16);
  v4f b1 = *(const v4f*)(p + kb + 20);
  v16h r;
#pragma unroll
  for (int i = 0; i < 4; ++i) {
    r[i]      = (_Float16)a0[i];
    r[i + 4]  = (_Float16)a1[i];
    r[i + 8]  = (_Float16)b0[i];
    r[i + 12] = (_Float16)b1[i];
  }
  return r;
}

// ---------------------------------------------------------------------------
__global__ void zero_kernel(float4* p, long n4) {
  long i = (long)blockIdx.x * blockDim.x + threadIdx.x;
  if (i < n4) p[i] = make_float4(0.f, 0.f, 0.f, 0.f);
}

__global__ void deg_kernel(const int* __restrict__ dst, float* __restrict__ deg) {
  long e = (long)blockIdx.x * blockDim.x + threadIdx.x;
  if (e < N_EDGES) atomicAdd(&deg[dst[e]], 1.0f);
}

__global__ void dinv_kernel(const float* __restrict__ deg, float* __restrict__ dinv) {
  long i = (long)blockIdx.x * blockDim.x + threadIdx.x;
  if (i < N_NODES) dinv[i] = rsqrtf(fmaxf(deg[i], 1.0f));
}

// ---------------------------------------------------------------------------
// Edge MLP + message formation + scatter-add.  One wave handles a 16-edge
// tile per loop iteration:
//   h1 = relu(edge_attr @ W1e + b1e)       (2 K-steps x 8 N-tiles WMMA)
//   msg = (h1 @ W2e + b2e + x[src]) * ew   (4 K-steps x 8 N-tiles WMMA)
//   atomicAdd agg[dst] += msg
// Weights staged to LDS once per block; persistent grid-stride over tiles.
// ---------------------------------------------------------------------------
__global__ void __launch_bounds__(128) edge_kernel(
    const float* __restrict__ x, const int* __restrict__ src,
    const int* __restrict__ dst, const float* __restrict__ ea,
    const float* __restrict__ W1e, const float* __restrict__ b1e,
    const float* __restrict__ W2e, const float* __restrict__ b2e,
    const float* __restrict__ dinv, float* __restrict__ agg) {
  __shared__ _Float16 sW1[128][64];    // W1e^T  (B-layout friendly)
  __shared__ _Float16 sW2[128][128];   // W2e^T
  __shared__ _Float16 sH[4][16][128];  // per-wave h1 staging (row-major f16)

  int t = threadIdx.x;
  for (int i = t; i < 64 * 128; i += 128) {
    int k = i >> 7, n = i & 127;
    sW1[n][k] = (_Float16)W1e[i];
  }
  for (int i = t; i < 128 * 128; i += 128) {
    int k = i >> 7, n = i & 127;
    sW2[n][k] = (_Float16)W2e[i];
  }
  __syncthreads();

  int wave = t >> 5, lane = t & 31;
  int row = lane & 15, hi = lane >> 4;
  int kb  = hi * 8;   // A-matrix K base per half
  int kbB = hi * 8;   // B-matrix K base per half
  _Float16* myH = &sH[wave][0][0];

  const int TILES  = N_EDGES / 16;            // 37500
  const int NSLOTS = EDGE_GRID * 4;
  int slot = blockIdx.x * 4 + wave;

  for (int tile = slot; tile < TILES; tile += NSLOTS) {
    long e0 = (long)tile * 16;

    // prefetch next iteration's edge_attr row for this lane
    if (tile + NSLOTS < TILES)
      __builtin_prefetch(ea + ((long)(tile + NSLOTS) * 16 + row) * EDIM, 0, 0);

    // per-lane metadata for its 8 C-layout rows (edges e0 + hi*8 + v)
    int se[8], de[8];
    float wt[8];
#pragma unroll
    for (int v = 0; v < 8; ++v) {
      long e = e0 + hi * 8 + v;
      int s = src[e], d = dst[e];
      se[v] = s;
      de[v] = d;
      wt[v] = dinv[s] * dinv[d];
    }

    // A operand for GEMM1: edge_attr tile 16x64, f32 -> f16
    const float* ap = ea + (e0 + row) * EDIM;
    v16h a1[2];
#pragma unroll
    for (int c = 0; c < 2; ++c) a1[c] = packA_f32(ap, 32 * c + kb);

    // GEMM1 + bias + relu -> LDS (f16 row-major)
#pragma unroll
    for (int nt = 0; nt < 8; ++nt) {
      int col = nt * 16 + row;
      v8f acc = {};
#pragma unroll
      for (int c = 0; c < 2; ++c) {
        v16h b = pack16(&sW1[col][32 * c + kbB], &sW1[col][32 * c + kbB + 16]);
        acc = wmma16(a1[c], b, acc);
      }
      float bias = b1e[col];
#pragma unroll
      for (int v = 0; v < 8; ++v)
        myH[(hi * 8 + v) * 128 + col] = (_Float16)fmaxf(acc[v] + bias, 0.0f);
    }
    asm volatile("s_wait_dscnt 0" ::: "memory");  // wave-local LDS RAW fence

    // A operand for GEMM2 (16x128 f16) read back in A-layout
    const _Float16* hp = myH + row * 128;
    v16h a2[4];
#pragma unroll
    for (int c = 0; c < 4; ++c)
      a2[c] = pack16(hp + 32 * c + kb, hp + 32 * c + kb + 16);

    // GEMM2 + bias + gather x[src] + scale + scatter to agg[dst]
#pragma unroll
    for (int nt = 0; nt < 8; ++nt) {
      int col = nt * 16 + row;
      v8f acc = {};
#pragma unroll
      for (int c = 0; c < 4; ++c) {
        v16h b = pack16(&sW2[col][32 * c + kbB], &sW2[col][32 * c + kbB + 16]);
        acc = wmma16(a2[c], b, acc);
      }
      float bias = b2e[col];
#pragma unroll
      for (int v = 0; v < 8; ++v) {
        float msg = (acc[v] + bias + x[(long)se[v] * DIM + col]) * wt[v];
        atomicAdd(&agg[(long)de[v] * DIM + col], msg);
      }
    }
    asm volatile("s_wait_dscnt 0" ::: "memory");  // WAR: a2 reads vs next store
  }
}

// ---------------------------------------------------------------------------
// Node MLP part 1: h1 = relu(((1+eps)*x + agg) @ W1n + b1n) -> ws (f16)
// ---------------------------------------------------------------------------
__global__ void __launch_bounds__(128) node_mlp1_kernel(
    const float* __restrict__ x, const float* __restrict__ agg,
    const float* __restrict__ W1n, const float* __restrict__ b1n,
    const float* __restrict__ geps_p, _Float16* __restrict__ h1) {
  __shared__ _Float16 sW[128][128];   // W1n^T
  __shared__ _Float16 sH[4][16][128]; // staging for coalesced b128 stores
  int t = threadIdx.x;
  for (int i = t; i < 128 * 128; i += 128) {
    int k = i >> 7, n = i & 127;
    sW[n][k] = (_Float16)W1n[i];
  }
  __syncthreads();

  int wave = t >> 5, lane = t & 31;
  int row = lane & 15, hi = lane >> 4;
  int kb = hi * 8, kbB = hi * 8;
  _Float16* myH = &sH[wave][0][0];
  float ge = 1.0f + geps_p[0];

  const int TILES  = N_NODES / 16;  // 6250 exact
  const int NSLOTS = NODE_GRID * 4;

  for (int tile = blockIdx.x * 4 + wave; tile < TILES; tile += NSLOTS) {
    long r0 = (long)tile * 16;

    if (tile + NSLOTS < TILES) {
      __builtin_prefetch(x + ((long)(tile + NSLOTS) * 16 + row) * DIM, 0, 0);
      __builtin_prefetch(agg + ((long)(tile + NSLOTS) * 16 + row) * DIM, 0, 0);
    }

    const float* xp = x + (r0 + row) * DIM;
    const float* gp = agg + (r0 + row) * DIM;
    v16h a[4];
#pragma unroll
    for (int c = 0; c < 4; ++c) {
      int kc = 32 * c + kb;
      v4f x0 = *(const v4f*)(xp + kc),      x1 = *(const v4f*)(xp + kc + 4);
      v4f x2 = *(const v4f*)(xp + kc + 16), x3 = *(const v4f*)(xp + kc + 20);
      v4f g0 = *(const v4f*)(gp + kc),      g1 = *(const v4f*)(gp + kc + 4);
      v4f g2 = *(const v4f*)(gp + kc + 16), g3 = *(const v4f*)(gp + kc + 20);
      v16h r;
#pragma unroll
      for (int i = 0; i < 4; ++i) {
        r[i]      = (_Float16)(ge * x0[i] + g0[i]);
        r[i + 4]  = (_Float16)(ge * x1[i] + g1[i]);
        r[i + 8]  = (_Float16)(ge * x2[i] + g2[i]);
        r[i + 12] = (_Float16)(ge * x3[i] + g3[i]);
      }
      a[c] = r;
    }

#pragma unroll
    for (int nt = 0; nt < 8; ++nt) {
      int col = nt * 16 + row;
      v8f acc = {};
#pragma unroll
      for (int c = 0; c < 4; ++c) {
        v16h b = pack16(&sW[col][32 * c + kbB], &sW[col][32 * c + kbB + 16]);
        acc = wmma16(a[c], b, acc);
      }
      float bias = b1n[col];
#pragma unroll
      for (int v = 0; v < 8; ++v)
        myH[(hi * 8 + v) * 128 + col] = (_Float16)fmaxf(acc[v] + bias, 0.0f);
    }
    asm volatile("s_wait_dscnt 0" ::: "memory");

    // coalesced LDS -> global copy of the 16x128 f16 tile
    _Float16* dp = h1 + r0 * DIM + lane * 64;
    const _Float16* sp = myH + lane * 64;
#pragma unroll
    for (int i = 0; i < 8; ++i) *(v8h*)(dp + i * 8) = *(const v8h*)(sp + i * 8);
    asm volatile("s_wait_dscnt 0" ::: "memory");  // WAR vs next iteration
  }
}

// ---------------------------------------------------------------------------
// Node MLP part 2: out = RMSNorm(h1 @ W2n + b2n + x) * rms_w
// RMSNorm fused via intra-half shfl_xor reductions (C-layout rows live in one
// 16-lane half of the wave).
// ---------------------------------------------------------------------------
__global__ void __launch_bounds__(128) node_mlp2_kernel(
    const float* __restrict__ x, const _Float16* __restrict__ h1,
    const float* __restrict__ W2n, const float* __restrict__ b2n,
    const float* __restrict__ rms_w, float* __restrict__ out) {
  __shared__ _Float16 sW[128][128];  // W2n^T
  int t = threadIdx.x;
  for (int i = t; i < 128 * 128; i += 128) {
    int k = i >> 7, n = i & 127;
    sW[n][k] = (_Float16)W2n[i];
  }
  __syncthreads();

  int wave = t >> 5, lane = t & 31;
  int row = lane & 15, hi = lane >> 4;
  int kb = hi * 8, kbB = hi * 8;

  const int TILES  = N_NODES / 16;  // 6250
  const int NSLOTS = NODE_GRID * 4;

  for (int tile = blockIdx.x * 4 + wave; tile < TILES; tile += NSLOTS) {
    long r0 = (long)tile * 16;

    if (tile + NSLOTS < TILES)
      __builtin_prefetch(h1 + ((long)(tile + NSLOTS) * 16 + row) * DIM, 0, 0);

    const _Float16* hp = h1 + (r0 + row) * DIM;
    v16h a[4];
#pragma unroll
    for (int c = 0; c < 4; ++c)
      a[c] = pack16(hp + 32 * c + kb, hp + 32 * c + kb + 16);

    v8f ov[8];
    float ssq[8];
#pragma unroll
    for (int v = 0; v < 8; ++v) ssq[v] = 0.0f;

#pragma unroll
    for (int nt = 0; nt < 8; ++nt) {
      int col = nt * 16 + row;
      v8f acc = {};
#pragma unroll
      for (int c = 0; c < 4; ++c) {
        v16h b = pack16(&sW[col][32 * c + kbB], &sW[col][32 * c + kbB + 16]);
        acc = wmma16(a[c], b, acc);
      }
      float bias = b2n[col];
#pragma unroll
      for (int v = 0; v < 8; ++v) {
        float o = acc[v] + bias + x[(r0 + hi * 8 + v) * DIM + col];
        acc[v] = o;
        ssq[v] += o * o;
      }
      ov[nt] = acc;
    }

    // reduce squared sums across the 16 lanes holding each row
#pragma unroll
    for (int m = 1; m < 16; m <<= 1) {
#pragma unroll
      for (int v = 0; v < 8; ++v) ssq[v] += __shfl_xor(ssq[v], m, 32);
    }
    float scale[8];
#pragma unroll
    for (int v = 0; v < 8; ++v)
      scale[v] = rsqrtf(ssq[v] * (1.0f / 128.0f) + RMS_EPS);

#pragma unroll
    for (int nt = 0; nt < 8; ++nt) {
      int col = nt * 16 + row;
      float w = rms_w[col];
#pragma unroll
      for (int v = 0; v < 8; ++v)
        out[(r0 + hi * 8 + v) * DIM + col] = ov[nt][v] * scale[v] * w;
    }
  }
}

// ---------------------------------------------------------------------------
extern "C" void kernel_launch(void* const* d_in, const int* in_sizes, int n_in,
                              void* d_out, int out_size, void* d_ws,
                              size_t ws_size, hipStream_t stream) {
  const float* x    = (const float*)d_in[0];
  const int*   ei   = (const int*)d_in[1];
  const float* ea   = (const float*)d_in[2];
  const float* W1n  = (const float*)d_in[3];
  const float* b1n  = (const float*)d_in[4];
  const float* W2n  = (const float*)d_in[5];
  const float* b2n  = (const float*)d_in[6];
  const float* W1e  = (const float*)d_in[7];
  const float* b1e  = (const float*)d_in[8];
  const float* W2e  = (const float*)d_in[9];
  const float* b2e  = (const float*)d_in[10];
  const float* geps = (const float*)d_in[11];
  const float* rmsw = (const float*)d_in[12];
  float* out = (float*)d_out;

  // workspace layout: agg[N*D] f32 | deg[N] f32 | dinv[N] f32 | h1[N*D] f16
  float* agg   = (float*)d_ws;
  float* deg   = agg + (size_t)N_NODES * DIM;
  float* dinv  = deg + N_NODES;
  _Float16* h1 = (_Float16*)(dinv + N_NODES);

  const int* src = ei;
  const int* dst = ei + N_EDGES;

  long nz4 = ((long)N_NODES * DIM + N_NODES) / 4;  // agg + deg, both zeroed
  zero_kernel<<<(int)((nz4 + 255) / 256), 256, 0, stream>>>((float4*)agg, nz4);
  deg_kernel<<<(N_EDGES + 255) / 256, 256, 0, stream>>>(dst, deg);
  dinv_kernel<<<(N_NODES + 255) / 256, 256, 0, stream>>>(deg, dinv);

  edge_kernel<<<EDGE_GRID, 128, 0, stream>>>(x, src, dst, ea, W1e, b1e, W2e,
                                             b2e, dinv, agg);
  node_mlp1_kernel<<<NODE_GRID, 128, 0, stream>>>(x, agg, W1n, b1n, geps, h1);
  node_mlp2_kernel<<<NODE_GRID, 128, 0, stream>>>(x, h1, W2n, b2n, rmsw, out);
}
